// GNOBlock_56057913147459
// MI455X (gfx1250) — matrix operand order repaired and checked by reading, not compiled, not run on hardware
//
#include <hip/hip_runtime.h>
#include <hip/hip_bf16.h>

typedef _Float16 h16;
typedef __attribute__((ext_vector_type(16))) _Float16 v16h;
typedef __attribute__((ext_vector_type(8)))  _Float16 v8h;
typedef __attribute__((ext_vector_type(8)))  float    v8f;

#define N_PTS   32768
#define M_PTS   32768
#define E_EDGES 524288
#define OUT_CH  32
#define EMB     192
#define WAVES   4

// Fast GELU: tanh form, using the gfx1250 hardware transcendental when the
// builtin exists (v_tanh_f32, co-executes with the WMMA/XDL pipe), else a
// sigmoid form on hardware v_exp_f32. ~7 VALU/TRANS ops vs ~30 for erff.
__device__ __forceinline__ float gelu_fast(float x) {
  float z = 0.7978845608028654f * x * (1.0f + 0.044715f * x * x);
#if __has_builtin(__builtin_amdgcn_tanhf)
  return 0.5f * x * (1.0f + __builtin_amdgcn_tanhf(z));
#else
  // 0.5*(1+tanh z) == sigmoid(2z); safe at both extremes.
  return x / (1.0f + __expf(-2.0f * z));
#endif
}

__device__ __forceinline__ v16h cat8(v8h lo, v8h hi) {
  return __builtin_shufflevector(lo, hi, 0,1,2,3,4,5,6,7,8,9,10,11,12,13,14,15);
}

// ---------------------------------------------------------------------------
// Positional embedding: emb[point][d*64 + 2k + {sin,cos}] as f16, row-major.
// freq_k = (1/10000)^(k/32) = exp(-k * ln(1e4)/32)
// ---------------------------------------------------------------------------
__global__ void emb_kernel(const float* __restrict__ y, const float* __restrict__ x,
                           h16* __restrict__ emb) {
  int tid = blockIdx.x * blockDim.x + threadIdx.x;   // (N+M)*96 threads
  if (tid >= (N_PTS + M_PTS) * 96) return;
  int point = tid / 96;
  int j     = tid - point * 96;   // j = d*32 + k
  int d = j >> 5;
  int k = j & 31;
  float p = (point < N_PTS) ? y[point * 3 + d] : x[(point - N_PTS) * 3 + d];
  float freq = __expf(-(float)k * 0.2878231366242557f);
  float ang  = p * freq;
  h16* dst = emb + (size_t)point * EMB + d * 64 + 2 * k;
  dst[0] = (h16)sinf(ang);
  dst[1] = (h16)cosf(ang);
}

// ---------------------------------------------------------------------------
// Weight transpose + f16 convert: src f32 [K][Nout] -> dst f16 [Nout][K]
// ---------------------------------------------------------------------------
__global__ void wtr_kernel(const float* __restrict__ src, h16* __restrict__ dst,
                           int K, int Nout) {
  int tid = blockIdx.x * blockDim.x + threadIdx.x;
  if (tid >= K * Nout) return;
  int k = tid / Nout;
  int n = tid - k * Nout;
  dst[(size_t)n * K + k] = (h16)src[(size_t)k * Nout + n];
}

__global__ void zero_kernel(float* __restrict__ p, int n) {
  int tid = blockIdx.x * blockDim.x + threadIdx.x;
  if (tid < n) p[tid] = 0.0f;
}

// ---------------------------------------------------------------------------
// Main: one wave = one 16-edge tile through the full MLP via WMMA f16->f32.
// ---------------------------------------------------------------------------
__global__ __launch_bounds__(WAVES * 32)
void gno_main(const h16* __restrict__ emb,
              const h16* __restrict__ w0t, const h16* __restrict__ w1t,
              const h16* __restrict__ w2t, const h16* __restrict__ w3t,
              const float* __restrict__ bb0, const float* __restrict__ bb1,
              const float* __restrict__ bb2, const float* __restrict__ bb3,
              const float* __restrict__ f_y,
              const int* __restrict__ nbr, const int* __restrict__ qry,
              float* __restrict__ out) {
  __shared__ __attribute__((aligned(32))) h16 sbuf0[WAVES][16 * 128]; // 16 KB
  __shared__ __attribute__((aligned(32))) h16 sbuf1[WAVES][16 * 256]; // 32 KB

  const int lane  = threadIdx.x & 31;
  const int wave  = threadIdx.x >> 5;
  const int tile  = blockIdx.x * WAVES + wave;
  const int ebase = tile * 16;
  const int m     = lane & 15;   // A row / B & D column within a 16-wide tile
  const int hsel  = lane >> 4;   // half-wave select
  const int drow  = hsel * 8;    // D-layout row offset
  const int abase = hsel * 16;   // A-fragment K base (elements)
  const int bbase = hsel * 16;   // B-fragment K base (elements)

  h16* bufA = &sbuf0[wave][0];   // 16 x 128 f16
  h16* bufB = &sbuf1[wave][0];   // 16 x 256 f16

  const int my_nbr = nbr[ebase + m];
  const int my_qry = qry[ebase + m];
  const h16* rowy = emb + (size_t)my_nbr * EMB;
  const h16* rowx = emb + (size_t)(N_PTS + my_qry) * EMB;

  // ---- Layer 0: [16x384] @ [384x128] + b0, GELU -> bufA --------------------
  {
    v8f acc[8];
#pragma unroll
    for (int t = 0; t < 8; ++t) {
      float bv = bb0[t * 16 + m];
#pragma unroll
      for (int r = 0; r < 8; ++r) acc[t][r] = bv;
    }
#pragma unroll
    for (int k0 = 0; k0 < 12; ++k0) {
      const h16* src = (k0 < 6) ? (rowy + k0 * 32) : (rowx + (k0 - 6) * 32);
      v16h a = cat8(*(const v8h*)(src + abase), *(const v8h*)(src + abase + 8));
#pragma unroll
      for (int t = 0; t < 8; ++t) {
        v16h b = *(const v16h*)(w0t + (size_t)(t * 16 + m) * 384 + k0 * 32 + bbase);
        acc[t] = __builtin_amdgcn_wmma_f32_16x16x32_f16(false, a, false, b,
                                                        (short)0, acc[t], false, false);
      }
    }
#pragma unroll
    for (int t = 0; t < 8; ++t)
#pragma unroll
      for (int r = 0; r < 8; ++r)
        bufA[(r + drow) * 128 + t * 16 + m] = (h16)gelu_fast(acc[t][r]);
  }

  // ---- Layer 1: [16x128] @ [128x256] + b1, GELU -> bufB (two N halves) -----
  for (int hh = 0; hh < 2; ++hh) {
    v8f acc[8];
#pragma unroll
    for (int t = 0; t < 8; ++t) {
      float bv = bb1[hh * 128 + t * 16 + m];
#pragma unroll
      for (int r = 0; r < 8; ++r) acc[t][r] = bv;
    }
#pragma unroll
    for (int k0 = 0; k0 < 4; ++k0) {
      const h16* src = bufA + m * 128 + k0 * 32;
      v16h a = cat8(*(const v8h*)(src + abase), *(const v8h*)(src + abase + 8));
#pragma unroll
      for (int t = 0; t < 8; ++t) {
        v16h b = *(const v16h*)(w1t + (size_t)(hh * 128 + t * 16 + m) * 128 + k0 * 32 + bbase);
        acc[t] = __builtin_amdgcn_wmma_f32_16x16x32_f16(false, a, false, b,
                                                        (short)0, acc[t], false, false);
      }
    }
#pragma unroll
    for (int t = 0; t < 8; ++t)
#pragma unroll
      for (int r = 0; r < 8; ++r)
        bufB[(r + drow) * 256 + hh * 128 + t * 16 + m] = (h16)gelu_fast(acc[t][r]);
  }

  // ---- Layer 2: [16x256] @ [256x128] + b2, GELU -> bufA --------------------
  {
    v8f acc[8];
#pragma unroll
    for (int t = 0; t < 8; ++t) {
      float bv = bb2[t * 16 + m];
#pragma unroll
      for (int r = 0; r < 8; ++r) acc[t][r] = bv;
    }
#pragma unroll
    for (int k0 = 0; k0 < 8; ++k0) {
      const h16* src = bufB + m * 256 + k0 * 32;
      v16h a = cat8(*(const v8h*)(src + abase), *(const v8h*)(src + abase + 8));
#pragma unroll
      for (int t = 0; t < 8; ++t) {
        v16h b = *(const v16h*)(w2t + (size_t)(t * 16 + m) * 256 + k0 * 32 + bbase);
        acc[t] = __builtin_amdgcn_wmma_f32_16x16x32_f16(false, a, false, b,
                                                        (short)0, acc[t], false, false);
      }
    }
#pragma unroll
    for (int t = 0; t < 8; ++t)
#pragma unroll
      for (int r = 0; r < 8; ++r)
        bufA[(r + drow) * 128 + t * 16 + m] = (h16)gelu_fast(acc[t][r]);
  }

  // ---- Layer 3: [16x128] @ [128x32] + b3 (no activation) -------------------
  v8f acc3[2];
#pragma unroll
  for (int t = 0; t < 2; ++t) {
    float bv = bb3[t * 16 + m];
#pragma unroll
    for (int r = 0; r < 8; ++r) acc3[t][r] = bv;
  }
#pragma unroll
  for (int k0 = 0; k0 < 4; ++k0) {
    const h16* src = bufA + m * 128 + k0 * 32;
    v16h a = cat8(*(const v8h*)(src + abase), *(const v8h*)(src + abase + 8));
#pragma unroll
    for (int t = 0; t < 2; ++t) {
      v16h b = *(const v16h*)(w3t + (size_t)(t * 16 + m) * 128 + k0 * 32 + bbase);
      acc3[t] = __builtin_amdgcn_wmma_f32_16x16x32_f16(false, a, false, b,
                                                       (short)0, acc3[t], false, false);
    }
  }

  // ---- rep = k * f_y[nbr], segment-sum by (sorted) query -------------------
  // Lane owns 8 consecutive edge rows at a fixed column -> run-length merge
  // equal queries before the global atomic.
  int qv[8], nv[8];
#pragma unroll
  for (int r = 0; r < 8; ++r) {
    int e = ebase + drow + r;
    qv[r] = qry[e];
    nv[r] = nbr[e];
  }
#pragma unroll
  for (int t = 0; t < 2; ++t) {
    const int col = t * 16 + m;
    float run  = 0.0f;
    int   curq = qv[0];
#pragma unroll
    for (int r = 0; r < 8; ++r) {
      float v = acc3[t][r] * f_y[(size_t)nv[r] * OUT_CH + col];
      if (qv[r] != curq) {
        atomicAdd(out + (size_t)curq * OUT_CH + col, run);
        run = 0.0f;
        curq = qv[r];
      }
      run += v;
    }
    atomicAdd(out + (size_t)curq * OUT_CH + col, run);
  }
}

// ---------------------------------------------------------------------------
extern "C" void kernel_launch(void* const* d_in, const int* in_sizes, int n_in,
                              void* d_out, int out_size, void* d_ws, size_t ws_size,
                              hipStream_t stream) {
  const float* y   = (const float*)d_in[0];
  const float* x   = (const float*)d_in[1];
  const float* f_y = (const float*)d_in[2];
  const int*   nbr = (const int*)d_in[3];
  const int*   qry = (const int*)d_in[4];
  const float* W0  = (const float*)d_in[5];
  const float* b0  = (const float*)d_in[6];
  const float* W1  = (const float*)d_in[7];
  const float* b1  = (const float*)d_in[8];
  const float* W2  = (const float*)d_in[9];
  const float* b2  = (const float*)d_in[10];
  const float* W3  = (const float*)d_in[11];
  const float* b3  = (const float*)d_in[12];
  float* out = (float*)d_out;

  char* ws = (char*)d_ws;
  h16* emb = (h16*)ws;
  size_t off = (size_t)(N_PTS + M_PTS) * EMB * sizeof(h16);
  h16* w0t = (h16*)(ws + off); off += (size_t)128 * 384 * sizeof(h16);
  h16* w1t = (h16*)(ws + off); off += (size_t)256 * 128 * sizeof(h16);
  h16* w2t = (h16*)(ws + off); off += (size_t)128 * 256 * sizeof(h16);
  h16* w3t = (h16*)(ws + off); off += (size_t)32  * 128 * sizeof(h16);

  // 1) embeddings (f16 tables, row-major [point][192])
  {
    int total = (N_PTS + M_PTS) * 96;
    emb_kernel<<<(total + 255) / 256, 256, 0, stream>>>(y, x, emb);
  }
  // 2) weight transpose + f16 convert
  wtr_kernel<<<(384 * 128 + 255) / 256, 256, 0, stream>>>(W0, w0t, 384, 128);
  wtr_kernel<<<(128 * 256 + 255) / 256, 256, 0, stream>>>(W1, w1t, 128, 256);
  wtr_kernel<<<(256 * 128 + 255) / 256, 256, 0, stream>>>(W2, w2t, 256, 128);
  wtr_kernel<<<(128 * 32  + 255) / 256, 256, 0, stream>>>(W3, w3t, 128, 32);
  // 3) zero the output (atomics accumulate into it)
  {
    int total = M_PTS * OUT_CH;
    zero_kernel<<<(total + 255) / 256, 256, 0, stream>>>(out, total);
  }
  // 4) main fused MLP + scatter
  {
    int tiles  = E_EDGES / 16;        // 32768
    int blocks = tiles / WAVES;       // 8192
    gno_main<<<blocks, WAVES * 32, 0, stream>>>(emb, w0t, w1t, w2t, w3t,
                                                b0, b1, b2, b3,
                                                f_y, nbr, qry, out);
  }
}
